// Sashimi_60550448939158
// MI455X (gfx1250) — matrix operand order repaired and checked by reading, not compiled, not run on hardware
//
#include <hip/hip_runtime.h>
#include <hip/hip_bf16.h>
#include <math.h>

// ---------------------------------------------------------------------------
// Sashimi forward for MI455X (gfx1250, wave32, WMMA).
// 1x1 linears run on v_wmma_f32_16x16x32_f16: weights pre-converted to f16
// (A fragments = 2x global b128 loads), activations staged column-major in
// LDS as f16 (B fragments = 2x ds_load_b128), 16x64 output strip per wave.
// The S4 FFT convolution is replaced by its exact diagonal-SSM recurrence
// (s = a*s + z, y = Re(G.s)), 8 complex states/lane + 8-lane shuffle reduce.
// ---------------------------------------------------------------------------

#define BATCH 4
#define NSTATE 64

typedef __attribute__((ext_vector_type(16))) _Float16 v16h;
typedef __attribute__((ext_vector_type(8)))  _Float16 v8h;
typedef __attribute__((ext_vector_type(8)))  float    v8f;

__device__ __forceinline__ float gelu_f(float v) {
    return 0.5f * v * (1.0f + erff(v * 0.70710678118654752f));
}
__device__ __forceinline__ float sigmoid_f(float v) {
    return __builtin_amdgcn_rcpf(1.0f + __expf(-v));
}

// ---------------- f32 -> f16 weight conversion ------------------------------
__global__ void cvt_f16(const float* __restrict__ in, _Float16* __restrict__ out,
                        int n) {
    int i = blockIdx.x * blockDim.x + threadIdx.x;
    if (i < n) out[i] = (_Float16)in[i];
}

// ---------------- WMMA A fragment from f16 weights --------------------------
// A 16x32 (MxK) layout: lanes 0-15 M, K-base 0; lanes 16-31 K-base 16;
// halfs 0..15 = K kb+0..15  (one contiguous run).
__device__ __forceinline__ v16h load_a16(const _Float16* __restrict__ W, int K,
                                         int row0, int k0, int lane) {
    int m  = lane & 15;
    int kb = (lane & 16) ? 16 : 0;
    const _Float16* r = W + (size_t)(row0 + m) * K + k0 + kb;
    v8h lo = *(const v8h*)r;          // global_load_b128
    v8h hi = *(const v8h*)(r + 8);    // global_load_b128
    return __builtin_shufflevector(lo, hi, 0, 1, 2, 3, 4, 5, 6, 7,
                                   8, 9, 10, 11, 12, 13, 14, 15);
}

// ---------------- B fragment from column-major LDS tile ---------------------
// xs[col][k]: 64 columns x 32 K-halfs, padded to 40 halfs (80 B, 16 B aligned,
// 20-bank stride => conflict free). Lanes 0-15: K 0..15; lanes 16-31: K 16..31.
__device__ __forceinline__ v16h load_b_lds(const _Float16 (*xs)[40], int sub,
                                           int lane) {
    const _Float16* cp = &xs[(sub << 4) + (lane & 15)][lane & 16];
    return *(const v16h*)cp;          // 2x ds_load_b128
}

// Cooperative stage of a 32(K) x 64(N) f32 slice into LDS (f16, column-major).
// Packs K-pairs into u32 stores; global reads coalesced across columns.
__device__ __forceinline__ void stage_tile(_Float16 (*xs)[40],
                                           const float* __restrict__ Xb,
                                           int k0, int nBase, int ld, int tid) {
#pragma unroll
    for (int i = 0; i < 8; ++i) {
        int p  = i * 128 + tid;       // 1024 K-pair slots
        int c  = p & 63;
        int kp = p >> 6;              // 0..15
        float v0 = Xb[(size_t)(k0 + 2 * kp) * ld + nBase + c];
        float v1 = Xb[(size_t)(k0 + 2 * kp + 1) * ld + nBase + c];
        union { _Float16 h[2]; unsigned int u; } pk;
        pk.h[0] = (_Float16)v0;
        pk.h[1] = (_Float16)v1;
        *(unsigned int*)&xs[c][2 * kp] = pk.u;
    }
}

// ---------------- Generic fused linear: Y = W*X + b  (per-batch K x Lc) -----
// EPI: 0 = bias, 1 = bias+GELU, 2 = bias + residual add (R).
// Block: 128 threads (4 waves); wave w computes rows [blockIdx.y*64 + 16w, +16)
// over a shared 64-column strip. Grid: (Lc/64, O/64, B).
template <int EPI>
__global__ __launch_bounds__(128) void linear_wmma(
    const _Float16* __restrict__ W, const float* __restrict__ bias,
    const float* __restrict__ X, float* __restrict__ Y,
    const float* __restrict__ R, int O, int K, int Lc) {
    __shared__ _Float16 xs[64][40];
    int lane  = threadIdx.x & 31;
    int wave  = threadIdx.x >> 5;
    int nBase = blockIdx.x * 64;
    int mTile = blockIdx.y * 64 + wave * 16;
    const float* Xb = X + (size_t)blockIdx.z * K * Lc;
    size_t boffY    = (size_t)blockIdx.z * O * Lc;

    v8f acc[4] = {v8f{}, v8f{}, v8f{}, v8f{}};
    for (int k0 = 0; k0 < K; k0 += 32) {
        __syncthreads();
        stage_tile(xs, Xb, k0, nBase, Lc, threadIdx.x);
        __syncthreads();
        v16h af = load_a16(W, K, mTile, k0, lane);
#pragma unroll
        for (int t = 0; t < 4; ++t) {
            v16h bf = load_b_lds(xs, t, lane);
            acc[t] = __builtin_amdgcn_wmma_f32_16x16x32_f16(
                false, af, false, bf, (short)0, acc[t], false, false);
        }
    }
    int rofs = (lane & 16) ? 8 : 0;
#pragma unroll
    for (int t = 0; t < 4; ++t) {
        int col = nBase + (t << 4) + (lane & 15);
#pragma unroll
        for (int r = 0; r < 8; ++r) {
            int   o = mTile + r + rofs;
            float v = acc[t][r] + bias[o];
            if (EPI == 1) v = gelu_f(v);
            size_t oi = boffY + (size_t)o * Lc + col;
            if (EPI == 2) v += R[oi];
            Y[oi] = v;
        }
    }
}

// ---------------- S4 output projection: x += GLU(Wo*y + bo) -----------------
// Both GLU halves share the staged B tile (8 WMMAs per staged tile).
__global__ __launch_bounds__(128) void glu_wmma(
    const _Float16* __restrict__ Wo, const float* __restrict__ bo,
    const float* __restrict__ Yact, float* __restrict__ Xio, int H, int Lc) {
    __shared__ _Float16 ys[64][40];
    int lane  = threadIdx.x & 31;
    int wave  = threadIdx.x >> 5;
    int nBase = blockIdx.x * 64;
    int mTile = blockIdx.y * 64 + wave * 16;
    const float* Yb = Yact + (size_t)blockIdx.z * H * Lc;
    size_t boffX    = (size_t)blockIdx.z * H * Lc;
    const _Float16* W2 = Wo + (size_t)H * H;   // rows H..2H-1 (gate half)

    v8f acc1[4] = {v8f{}, v8f{}, v8f{}, v8f{}};
    v8f acc2[4] = {v8f{}, v8f{}, v8f{}, v8f{}};
    for (int k0 = 0; k0 < H; k0 += 32) {
        __syncthreads();
        stage_tile(ys, Yb, k0, nBase, Lc, threadIdx.x);
        __syncthreads();
        v16h a1 = load_a16(Wo, H, mTile, k0, lane);
        v16h a2 = load_a16(W2, H, mTile, k0, lane);
#pragma unroll
        for (int t = 0; t < 4; ++t) {
            v16h bf = load_b_lds(ys, t, lane);
            acc1[t] = __builtin_amdgcn_wmma_f32_16x16x32_f16(
                false, a1, false, bf, (short)0, acc1[t], false, false);
            acc2[t] = __builtin_amdgcn_wmma_f32_16x16x32_f16(
                false, a2, false, bf, (short)0, acc2[t], false, false);
        }
    }
    int rofs = (lane & 16) ? 8 : 0;
#pragma unroll
    for (int t = 0; t < 4; ++t) {
        int col = nBase + (t << 4) + (lane & 15);
#pragma unroll
        for (int r = 0; r < 8; ++r) {
            int   o  = mTile + r + rofs;
            float v1 = acc1[t][r] + bo[o];
            float v2 = acc2[t][r] + bo[H + o];
            size_t oi = boffX + (size_t)o * Lc + col;
            Xio[oi] += v1 * sigmoid_f(v2);
        }
    }
}

// ---------------- LayerNorm over channel axis of (B,H,Lc) -------------------
__global__ __launch_bounds__(256) void ln_ch(const float* __restrict__ x,
                                             const float* __restrict__ g,
                                             const float* __restrict__ bt,
                                             float* __restrict__ z, int H, int Lc) {
    int wid  = threadIdx.x >> 5;
    int lane = threadIdx.x & 31;
    int pos  = blockIdx.x * 8 + wid;       // b*Lc + l
    int b = pos / Lc, l = pos - b * Lc;
    const float* xb = x + (size_t)b * H * Lc + l;
    int nC = H >> 5;
    float v[8];
    float s = 0.f, s2 = 0.f;
#pragma unroll
    for (int i = 0; i < 8; ++i) {
        if (i < nC) {
            int c = lane + (i << 5);
            v[i] = xb[(size_t)c * Lc];
            s += v[i];
            s2 += v[i] * v[i];
        }
    }
#pragma unroll
    for (int off = 16; off >= 1; off >>= 1) {
        s  += __shfl_xor(s, off, 32);
        s2 += __shfl_xor(s2, off, 32);
    }
    float m  = s / (float)H;
    float vr = s2 / (float)H - m * m;
    float rs = rsqrtf(vr + 1e-5f);
#pragma unroll
    for (int i = 0; i < 8; ++i) {
        if (i < nC) {
            int c = lane + (i << 5);
            z[(size_t)(b * H + c) * Lc + l] = (v[i] - m) * rs * g[c] + bt[c];
        }
    }
}

// ---------------- Final LayerNorm over last dim + transpose to (B,L,H) ------
__global__ __launch_bounds__(256) void ln_last(const float* __restrict__ x,
                                               const float* __restrict__ g,
                                               const float* __restrict__ bt,
                                               float* __restrict__ out,
                                               int H, int Lc) {
    int wid  = threadIdx.x >> 5;
    int lane = threadIdx.x & 31;
    int pos  = blockIdx.x * 8 + wid;
    int b = pos / Lc, l = pos - b * Lc;
    const float* xb = x + (size_t)b * H * Lc + l;
    float v[2];
    float s = 0.f, s2 = 0.f;
#pragma unroll
    for (int i = 0; i < 2; ++i) {
        int c = lane + (i << 5);
        v[i] = xb[(size_t)c * Lc];
        s += v[i];
        s2 += v[i] * v[i];
    }
#pragma unroll
    for (int off = 16; off >= 1; off >>= 1) {
        s  += __shfl_xor(s, off, 32);
        s2 += __shfl_xor(s2, off, 32);
    }
    float m  = s / (float)H;
    float vr = s2 / (float)H - m * m;
    float rs = rsqrtf(vr + 1e-5f);
#pragma unroll
    for (int i = 0; i < 2; ++i) {
        int c = lane + (i << 5);
        out[((size_t)b * Lc + l) * H + c] = (v[i] - m) * rs * g[c] + bt[c];
    }
}

// ---------------- (B,L,H) -> (B,H,L) transpose ------------------------------
__global__ void transpose_blh(const float* __restrict__ in, float* __restrict__ out,
                              int Lc, int H) {
    __shared__ float tile[16][17];
    int b = blockIdx.z, l0 = blockIdx.x * 16, h0 = blockIdx.y * 16;
    tile[threadIdx.y][threadIdx.x] =
        in[((size_t)b * Lc + l0 + threadIdx.y) * H + h0 + threadIdx.x];
    __syncthreads();
    out[((size_t)b * H + h0 + threadIdx.y) * Lc + l0 + threadIdx.x] =
        tile[threadIdx.x][threadIdx.y];
}

// ---------------- DownPool rearrange: (B,H,L) -> (B,H*s,L/s) ----------------
__global__ void rearrange_down(const float* __restrict__ x, float* __restrict__ out,
                               int Hin, int Lin, int s, int total) {
    int idx = blockIdx.x * blockDim.x + threadIdx.x;
    if (idx >= total) return;
    int Lq = Lin / s, Hs = Hin * s;
    int lq = idx % Lq;
    int c  = (idx / Lq) % Hs;
    int b  = idx / (Lq * Hs);
    out[idx] = x[((size_t)b * Hin + c / s) * Lin + lq * s + (c % s)];
}

// ---------------- UpPool: causal shift + expand + skip add ------------------
__global__ void up_expand(const float* __restrict__ zt, const float* __restrict__ skip,
                          float* __restrict__ xout, float* __restrict__ skipout,
                          int Ho, int Lc, int s, int total) {
    int idx = blockIdx.x * blockDim.x + threadIdx.x;
    if (idx >= total) return;
    int Lb = Lc * s;
    int lb = idx % Lb;
    int ho = (idx / Lb) % Ho;
    int b  = idx / (Lb * Ho);
    int l = lb / s, j = lb % s;
    float zv = (l > 0) ? zt[((size_t)b * Ho * s + ho * s + j) * Lc + (l - 1)] : 0.f;
    float v  = zv + skip[idx];
    xout[idx]    = v;
    skipout[idx] = v;
}

__global__ void add_inplace(float* __restrict__ x, const float* __restrict__ y, int n) {
    int i = blockIdx.x * blockDim.x + threadIdx.x;
    if (i < n) x[i] += y[i];
}

// ---------------- S4 discretization: a = exp(dt*A), G = 2*C*B*(a-1)/A -------
__global__ void s4_prep(const float* __restrict__ log_dt,
                        const float* __restrict__ A_re, const float* __restrict__ A_im,
                        const float* __restrict__ B_re, const float* __restrict__ B_im,
                        const float* __restrict__ C_re, const float* __restrict__ C_im,
                        float* __restrict__ a_re, float* __restrict__ a_im,
                        float* __restrict__ g_re, float* __restrict__ g_im,
                        int H, int N) {
    int t = blockIdx.x * blockDim.x + threadIdx.x;
    if (t >= H * N) return;
    int h = t / N;
    float dt = expf(log_dt[h]);
    float ar = A_re[t], ai = A_im[t];
    float er = expf(dt * ar);
    float ec = er * cosf(dt * ai);
    float es = er * sinf(dt * ai);
    float m1r = ec - 1.f, m1i = es;
    float den  = ar * ar + ai * ai;
    float ivr  = ar / den, ivi = -ai / den;      // 1/A = conj(A)/|A|^2
    float dBr = m1r * ivr - m1i * ivi;
    float dBi = m1r * ivi + m1i * ivr;
    float br = B_re[t], bi = B_im[t];
    float tr = dBr * br - dBi * bi;
    float ti = dBr * bi + dBi * br;
    float cr = C_re[t], ci = C_im[t];
    a_re[t] = ec;
    a_im[t] = es;
    g_re[t] = 2.f * (cr * tr - ci * ti);
    g_im[t] = 2.f * (cr * ti + ci * tr);
}

// ---------------- Diagonal SSM recurrence + D-skip + GELU -------------------
// 8 complex states per lane, 8 lanes per (b,h) channel, 4 channels per wave.
__global__ __launch_bounds__(256) void s4_ssm(
    const float* __restrict__ a_re, const float* __restrict__ a_im,
    const float* __restrict__ g_re, const float* __restrict__ g_im,
    const float* __restrict__ Dp, const float* __restrict__ z,
    float* __restrict__ y, int H, int Lc) {
    int grp  = threadIdx.x >> 3;           // channel within block (0..31)
    int lane = threadIdx.x & 7;            // lane within 8-lane group
    int ch = blockIdx.x * 32 + grp;        // b*H + h
    int h  = ch % H;
    const float* zb = z + (size_t)ch * Lc;
    float*       yb = y + (size_t)ch * Lc;

    float ar[8], ai[8], gr[8], gi[8], sr[8], si[8];
    int nb = h * NSTATE + lane * 8;
#pragma unroll
    for (int i = 0; i < 8; ++i) {
        ar[i] = a_re[nb + i]; ai[i] = a_im[nb + i];
        gr[i] = g_re[nb + i]; gi[i] = g_im[nb + i];
        sr[i] = 0.f; si[i] = 0.f;
    }
    float dcoef = Dp[h];

    for (int l = 0; l < Lc; l += 4) {
        float4 z4 = *(const float4*)(zb + l);
        __builtin_prefetch(zb + l + 64, 0, 0);     // global_prefetch_b8
        float out[4];
#pragma unroll
        for (int u = 0; u < 4; ++u) {
            float zv = (&z4.x)[u];
            float acc = 0.f;
#pragma unroll
            for (int i = 0; i < 8; ++i) {
                float nsr = ar[i] * sr[i] - ai[i] * si[i] + zv;
                si[i] = ar[i] * si[i] + ai[i] * sr[i];
                sr[i] = nsr;
                acc += gr[i] * sr[i] - gi[i] * si[i];
            }
            acc += __shfl_xor(acc, 1, 8);
            acc += __shfl_xor(acc, 2, 8);
            acc += __shfl_xor(acc, 4, 8);
            out[u] = gelu_f(acc + dcoef * zv);
        }
        if (lane == 0) *(float4*)(yb + l) = make_float4(out[0], out[1], out[2], out[3]);
    }
}

// ===========================================================================
// Host orchestration
// ===========================================================================
struct S4P { const float *ln_g,*ln_b,*log_dt,*A_re,*A_im,*B_re,*B_im,*C_re,*C_im,*D,*Wo,*bo; };
struct FFP { const float *ln_g,*ln_b,*W1,*b1,*W2,*b2; };

static void cvt_w(const float* w, _Float16* w16, int n, hipStream_t st) {
    cvt_f16<<<(n + 255) / 256, 256, 0, st>>>(w, w16, n);
}

static void run_block_pair(const S4P& sp, const FFP& fp, float* x, int H, int Lc,
                           float* z, float* y, float* tmp, float* par,
                           _Float16* w16, hipStream_t st) {
    ln_ch<<<BATCH * Lc / 8, 256, 0, st>>>(x, sp.ln_g, sp.ln_b, z, H, Lc);
    int HN = H * NSTATE;
    s4_prep<<<(HN + 255) / 256, 256, 0, st>>>(sp.log_dt, sp.A_re, sp.A_im,
                                              sp.B_re, sp.B_im, sp.C_re, sp.C_im,
                                              par, par + HN, par + 2 * HN, par + 3 * HN,
                                              H, NSTATE);
    s4_ssm<<<BATCH * H / 32, 256, 0, st>>>(par, par + HN, par + 2 * HN, par + 3 * HN,
                                           sp.D, z, y, H, Lc);
    cvt_w(sp.Wo, w16, 2 * H * H, st);
    glu_wmma<<<dim3(Lc / 64, H / 64, BATCH), 128, 0, st>>>(w16, sp.bo, y, x, H, Lc);
    // FF block
    ln_ch<<<BATCH * Lc / 8, 256, 0, st>>>(x, fp.ln_g, fp.ln_b, z, H, Lc);
    cvt_w(fp.W1, w16, 2 * H * H, st);
    linear_wmma<1><<<dim3(Lc / 64, 2 * H / 64, BATCH), 128, 0, st>>>(
        w16, fp.b1, z, tmp, nullptr, 2 * H, H, Lc);
    cvt_w(fp.W2, w16, 2 * H * H, st);
    linear_wmma<2><<<dim3(Lc / 64, H / 64, BATCH), 128, 0, st>>>(
        w16, fp.b2, tmp, x, x, H, 2 * H, Lc);
}

extern "C" void kernel_launch(void* const* d_in, const int* in_sizes, int n_in,
                              void* d_out, int out_size, void* d_ws, size_t ws_size,
                              hipStream_t stream) {
    (void)in_sizes; (void)n_in; (void)out_size; (void)ws_size;

    // ---- walk inputs in setup_inputs() insertion order ----
    int ii = 0;
    auto nx = [&]() { return (const float*)d_in[ii++]; };
    auto rdS4 = [&]() {
        S4P p;
        p.ln_g = nx(); p.ln_b = nx(); p.log_dt = nx();
        p.A_re = nx(); p.A_im = nx(); p.B_re = nx(); p.B_im = nx();
        p.C_re = nx(); p.C_im = nx(); p.D = nx(); p.Wo = nx(); p.bo = nx();
        return p;
    };
    auto rdFF = [&]() {
        FFP p;
        p.ln_g = nx(); p.ln_b = nx(); p.W1 = nx(); p.b1 = nx();
        p.W2 = nx(); p.b2 = nx();
        return p;
    };

    const float* xin = nx();                       // (B,L,64)
    const float *d0w = nx(), *d0b = nx();          // down[0]: (128,256)
    const float *d1w = nx(), *d1b = nx();          // down[1]: (256,512)
    S4P cs4[4]; FFP cff[4];
    for (int i = 0; i < 4; ++i) { cs4[i] = rdS4(); cff[i] = rdFF(); }
    const float *u0w = nx(), *u0b = nx();          // up[0].up: (512,256)
    S4P us4a[4]; FFP uffa[4];
    for (int i = 0; i < 4; ++i) { us4a[i] = rdS4(); uffa[i] = rdFF(); }
    const float *u1w = nx(), *u1b = nx();          // up[1].up: (256,128)
    S4P us4b[4]; FFP uffb[4];
    for (int i = 0; i < 4; ++i) { us4b[i] = rdS4(); uffb[i] = rdFF(); }
    const float *ng = nx(), *nb = nx();            // final norm

    // ---- workspace bump allocator ----
    size_t off = 0;
    char* base = (char*)d_ws;
    auto alloc_b = [&](size_t bytes) -> void* {
        void* p = base + off;
        off += (bytes + 255) & ~(size_t)255;
        return p;
    };
    auto allocf = [&](size_t elems) -> float* {
        return (float*)alloc_b(elems * sizeof(float));
    };
    float* x0  = allocf((size_t)BATCH * 64 * 4096);    // tier-0 tensor / skip
    float* x1  = allocf((size_t)BATCH * 128 * 1024);   // tier-1 tensor / skip
    float* x2  = allocf((size_t)BATCH * 256 * 256);    // center tensor
    float* sv  = allocf((size_t)BATCH * 256 * 256);    // center skip copy
    float* zb  = allocf((size_t)BATCH * 64 * 4096);    // LN output
    float* yb  = allocf((size_t)BATCH * 64 * 4096);    // SSM output
    float* tmp = allocf((size_t)BATCH * 128 * 4096);   // GEMM/rearrange scratch
    float* skp = allocf((size_t)BATCH * 64 * 4096);    // up-tier skip copy
    float* par = allocf((size_t)256 * NSTATE * 4);     // a_re,a_im,G_re,G_im
    _Float16* w16 = (_Float16*)alloc_b((size_t)512 * 256 * sizeof(_Float16));

    // ---- input transpose (B,L,H)->(B,H,L) ----
    transpose_blh<<<dim3(4096 / 16, 64 / 16, BATCH), dim3(16, 16), 0, stream>>>(
        xin, x0, 4096, 64);

    // ---- down path ----
    {
        int tot = BATCH * 256 * 1024;
        rearrange_down<<<tot / 256, 256, 0, stream>>>(x0, tmp, 64, 4096, 4, tot);
        cvt_w(d0w, w16, 128 * 256, stream);
        linear_wmma<0><<<dim3(1024 / 64, 128 / 64, BATCH), 128, 0, stream>>>(
            w16, d0b, tmp, x1, nullptr, 128, 256, 1024);
        tot = BATCH * 512 * 256;
        rearrange_down<<<tot / 256, 256, 0, stream>>>(x1, tmp, 128, 1024, 4, tot);
        cvt_w(d1w, w16, 256 * 512, stream);
        linear_wmma<0><<<dim3(256 / 64, 256 / 64, BATCH), 128, 0, stream>>>(
            w16, d1b, tmp, x2, nullptr, 256, 512, 256);
    }

    // ---- center blocks (H=256, L=256) + skip of deepest down output ----
    hipMemcpyAsync(sv, x2, (size_t)BATCH * 256 * 256 * sizeof(float),
                   hipMemcpyDeviceToDevice, stream);
    for (int i = 0; i < 4; ++i)
        run_block_pair(cs4[i], cff[i], x2, 256, 256, zb, yb, tmp, par, w16, stream);
    {
        int n = BATCH * 256 * 256;
        add_inplace<<<(n + 255) / 256, 256, 0, stream>>>(x2, sv, n);
    }

    // ---- up tier 0: 256 -> 128, L 256 -> 1024 ----
    cvt_w(u0w, w16, 512 * 256, stream);
    linear_wmma<0><<<dim3(256 / 64, 512 / 64, BATCH), 128, 0, stream>>>(
        w16, u0b, x2, tmp, nullptr, 512, 256, 256);
    {
        int tot = BATCH * 128 * 1024;
        up_expand<<<tot / 256, 256, 0, stream>>>(tmp, x1, x1, skp, 128, 256, 4, tot);
        for (int i = 0; i < 4; ++i)
            run_block_pair(us4a[i], uffa[i], x1, 128, 1024, zb, yb, tmp, par, w16, stream);
        add_inplace<<<tot / 256, 256, 0, stream>>>(x1, skp, tot);
    }

    // ---- up tier 1: 128 -> 64, L 1024 -> 4096 ----
    cvt_w(u1w, w16, 256 * 128, stream);
    linear_wmma<0><<<dim3(1024 / 64, 256 / 64, BATCH), 128, 0, stream>>>(
        w16, u1b, x1, tmp, nullptr, 256, 128, 1024);
    {
        int tot = BATCH * 64 * 4096;
        up_expand<<<tot / 256, 256, 0, stream>>>(tmp, x0, x0, skp, 64, 1024, 4, tot);
        for (int i = 0; i < 4; ++i)
            run_block_pair(us4b[i], uffb[i], x0, 64, 4096, zb, yb, tmp, par, w16, stream);
        add_inplace<<<tot / 256, 256, 0, stream>>>(x0, skp, tot);
    }

    // ---- final LayerNorm over channels, emit (B,L,H) ----
    ln_last<<<BATCH * 4096 / 8, 256, 0, stream>>>(x0, ng, nb, (float*)d_out, 64, 4096);
}
